// Attention_42185168781604
// MI455X (gfx1250) — matrix operand order repaired and checked by reading, not compiled, not run on hardware
//
#include <hip/hip_runtime.h>

typedef __attribute__((ext_vector_type(16))) _Float16 v16h;
typedef __attribute__((ext_vector_type(8)))  _Float16 v8h;
typedef __attribute__((ext_vector_type(8)))  float    v8f;

#define N_DIM 8192
#define M_DIM 8192
#define K_DIM 1024
#define H_DIM 1024
#define NCHUNK 16       // stats row-chunks
#define CHUNK  512      // rows per stats chunk

// ---- WMMA helpers -------------------------------------------------------
__device__ __forceinline__ v8f wmma_f16(v16h a, v16h b, v8f c) {
  // 8 args: (neg_a, A, neg_b, B, c_mod, C, reuse_a, reuse_b)
  return __builtin_amdgcn_wmma_f32_16x16x32_f16(false, a, false, b, (short)0, c, false, false);
}

// A-matrix 16x32 f16 fragment. Lane l holds row (rowbase + (l&15)).
// elements 0..7  = A[row, k + 8*hi + 0..7]
// elements 8..15 = A[row, k + 16 + 8*hi + 0..7]     (hi = l>>4)
__device__ __forceinline__ v16h ldfragA(const _Float16* a, int ld, int rowbase, int k, int lane) {
  int hi = lane >> 4;
  const _Float16* p = a + (size_t)(rowbase + (lane & 15)) * ld + k + hi * 8;
  v8h lo = *(const v8h*)p;
  v8h up = *(const v8h*)(p + 16);
  v16h r;
#pragma unroll
  for (int i = 0; i < 8; ++i) { r[i] = lo[i]; r[i + 8] = up[i]; }
  return r;
}

// B-matrix 32x16 f16 fragment, loaded from B^T (row-major N x K).
// Lane l holds column (colbase + (l&15)); elements 0..7 = Bt[col, k + 8*hi + 0..7], elements 8..15 = Bt[col, k + 16 + 8*hi + 0..7].
__device__ __forceinline__ v16h ldfragB(const _Float16* bt, int ld, int colbase, int k, int lane) {
  int hi = lane >> 4;
  const _Float16* p = bt + (size_t)(colbase + (lane & 15)) * ld + k + hi * 8;
  v8h lo = *(const v8h*)p;
  v8h up = *(const v8h*)(p + 16);
  v16h r;
#pragma unroll
  for (int i = 0; i < 8; ++i) { r[i] = lo[i]; r[i + 8] = up[i]; }
  return r;
}

// ---- prep kernels -------------------------------------------------------
__global__ void k_cvt(const float* __restrict__ in, _Float16* __restrict__ out, int n) {
  int i = blockIdx.x * blockDim.x + threadIdx.x;
  if (i < n) out[i] = (_Float16)in[i];
}

// out[c][r] = in[r][c] * (scale ? scale[r] : 1), f32 -> f16, tiled transpose
__global__ void k_transpose(const float* __restrict__ in, _Float16* __restrict__ out,
                            int R, int C, const float* __restrict__ scale) {
  __shared__ float tile[32][33];
  __shared__ float srow[32];
  int r0 = blockIdx.x * 32, c0 = blockIdx.y * 32;
  for (int i = threadIdx.y; i < 32; i += 8)
    tile[i][threadIdx.x] = in[(size_t)(r0 + i) * C + c0 + threadIdx.x];
  if (scale && threadIdx.y == 0) srow[threadIdx.x] = scale[r0 + threadIdx.x];
  __syncthreads();
  for (int i = threadIdx.y; i < 32; i += 8) {
    float v = tile[threadIdx.x][i];
    if (scale) v *= srow[threadIdx.x];
    out[(size_t)(c0 + i) * R + r0 + threadIdx.x] = (_Float16)v;
  }
}

// ---- GEMM: Q = y @ W  (A = y_h, Bt = Wt_h), Q in f16 --------------------
__global__ __launch_bounds__(256) void k_gemm_q(const _Float16* __restrict__ A,
                                                const _Float16* __restrict__ Bt,
                                                _Float16* __restrict__ Q) {
  int lane = threadIdx.x & 31, w = threadIdx.x >> 5;
  int n0 = blockIdx.x * 128 + (w & 3) * 32;   // 32 rows per wave
  int j0 = blockIdx.y * 128 + (w >> 2) * 64;  // 64 cols per wave
  v8f acc[2][4] = {};
  for (int k = 0; k < K_DIM; k += 32) {
    v16h a0 = ldfragA(A, K_DIM, n0, k, lane);
    v16h a1 = ldfragA(A, K_DIM, n0 + 16, k, lane);
    v16h b[4];
#pragma unroll
    for (int j = 0; j < 4; ++j) b[j] = ldfragB(Bt, K_DIM, j0 + j * 16, k, lane);
#pragma unroll
    for (int j = 0; j < 4; ++j) {
      acc[0][j] = wmma_f16(a0, b[j], acc[0][j]);
      acc[1][j] = wmma_f16(a1, b[j], acc[1][j]);
    }
  }
  int hi = lane >> 4, c = lane & 15;
#pragma unroll
  for (int i = 0; i < 2; ++i)
#pragma unroll
    for (int j = 0; j < 4; ++j)
#pragma unroll
      for (int v = 0; v < 8; ++v) {
        int row = n0 + i * 16 + hi * 8 + v;
        int col = j0 + j * 16 + c;
        Q[(size_t)row * H_DIM + col] = (_Float16)acc[i][j][v];
      }
}

// ---- stats: per-column (m) online max & sum of exp over a 512-row chunk --
__global__ __launch_bounds__(256) void k_stats(const _Float16* __restrict__ Q,
                                               const _Float16* __restrict__ Hh,
                                               float* __restrict__ pmax,
                                               float* __restrict__ psum) {
  int lane = threadIdx.x & 31, w = threadIdx.x >> 5;
  int m0 = blockIdx.x * 128 + w * 16;     // wave owns 16 columns
  int nb = blockIdx.y * CHUNK;
  float rmax = -3.0e38f, rsum = 0.0f;
  for (int rs = 0; rs < CHUNK / 32; ++rs) {
    int nr = nb + rs * 32;
    v8f s0 = {}, s1 = {};
    for (int k = 0; k < K_DIM; k += 32) {
      v16h a0 = ldfragA(Q, K_DIM, nr, k, lane);
      v16h a1 = ldfragA(Q, K_DIM, nr + 16, k, lane);
      v16h b  = ldfragB(Hh, K_DIM, m0, k, lane);
      s0 = wmma_f16(a0, b, s0);
      s1 = wmma_f16(a1, b, s1);
    }
    float mx = rmax;
#pragma unroll
    for (int v = 0; v < 8; ++v) mx = fmaxf(mx, fmaxf(s0[v], s1[v]));
    float acc = rsum * __expf(rmax - mx);
#pragma unroll
    for (int v = 0; v < 8; ++v) acc += __expf(s0[v] - mx) + __expf(s1[v] - mx);
    rmax = mx; rsum = acc;
  }
  // lanes l and l+16 hold the same column (different row halves) -> combine
  float om = __shfl_xor(rmax, 16, 32);
  float os = __shfl_xor(rsum, 16, 32);
  float fm = fmaxf(rmax, om);
  float fs = rsum * __expf(rmax - fm) + os * __expf(om - fm);
  if (lane < 16) {
    size_t idx = (size_t)blockIdx.y * M_DIM + m0 + lane;
    pmax[idx] = fm;
    psum[idx] = fs;
  }
}

__global__ void k_reduce(const float* __restrict__ pmax, const float* __restrict__ psum,
                         float* __restrict__ Mb, float* __restrict__ Dinv) {
  int m = blockIdx.x * blockDim.x + threadIdx.x;
  if (m >= M_DIM) return;
  float fm = -3.0e38f;
  for (int c = 0; c < NCHUNK; ++c) fm = fmaxf(fm, pmax[(size_t)c * M_DIM + m]);
  float s = 0.0f;
  for (int c = 0; c < NCHUNK; ++c)
    s += psum[(size_t)c * M_DIM + m] * __expf(pmax[(size_t)c * M_DIM + m] - fm);
  Mb[m] = fm;
  Dinv[m] = 1.0f / s;
}

// ---- fused final: out[n,:] = sum_m exp(s[n,m]-M[m]) * (h[m,:]/D[m]) -----
// Block: 64 rows x 1024 cols, 8 waves; per 32-wide m-chunk:
//   phase A: 8 waves recompute the 64x32 score tile (1 WMMA-tile each),
//            exp -> f16 P tile staged in LDS.
//   phase B: P (A-frags from LDS) x hdt (B-frags) accumulate 4x8 out tiles.
__global__ __launch_bounds__(256) void k_final(const _Float16* __restrict__ Q,
                                               const _Float16* __restrict__ Hh,
                                               const _Float16* __restrict__ HDt,
                                               const float* __restrict__ Mb,
                                               float* __restrict__ out) {
  __shared__ __align__(32) _Float16 plds[64 * 32];
  int lane = threadIdx.x & 31, w = threadIdx.x >> 5;
  int hi = lane >> 4, c = lane & 15;
  int n0 = blockIdx.x * 64;
  int jw = w * 128;                 // wave's 128 output columns
  int nsub = w & 3, msub = w >> 2;  // wave's score sub-tile in phase A
  v8f acc[4][8] = {};

  for (int mc = 0; mc < M_DIM; mc += 32) {
    // ---- phase A: recompute score sub-tile, write exp() to LDS ----
    v8f s = {};
    for (int k = 0; k < K_DIM; k += 32) {
      v16h a = ldfragA(Q,  K_DIM, n0 + nsub * 16, k, lane);
      v16h b = ldfragB(Hh, K_DIM, mc + msub * 16, k, lane);
      s = wmma_f16(a, b, s);
    }
    float Mv = Mb[mc + msub * 16 + c];
#pragma unroll
    for (int v = 0; v < 8; ++v) {
      float pv = __expf(s[v] - Mv);
      plds[(nsub * 16 + hi * 8 + v) * 32 + msub * 16 + c] = (_Float16)pv;
    }
    __syncthreads();

    // ---- phase B: out += P @ hdt over this 32-wide m-chunk ----
    v16h af[4];
#pragma unroll
    for (int it = 0; it < 4; ++it) {
      const _Float16* pr = &plds[(it * 16 + c) * 32 + hi * 8];
      v8h lo = *(const v8h*)pr;
      v8h up = *(const v8h*)(pr + 16);
      v16h r;
#pragma unroll
      for (int i = 0; i < 8; ++i) { r[i] = lo[i]; r[i + 8] = up[i]; }
      af[it] = r;
    }
#pragma unroll
    for (int jt = 0; jt < 8; ++jt) {
      v16h b = ldfragB(HDt, M_DIM, jw + jt * 16, mc, lane);
#pragma unroll
      for (int it = 0; it < 4; ++it) acc[it][jt] = wmma_f16(af[it], b, acc[it][jt]);
    }
    __syncthreads();
  }

#pragma unroll
  for (int it = 0; it < 4; ++it)
#pragma unroll
    for (int jt = 0; jt < 8; ++jt)
#pragma unroll
      for (int v = 0; v < 8; ++v) {
        int row = n0 + it * 16 + hi * 8 + v;
        int col = jw + jt * 16 + c;
        out[(size_t)row * H_DIM + col] = acc[it][jt][v];
      }
}

// ---- host launcher ------------------------------------------------------
extern "C" void kernel_launch(void* const* d_in, const int* in_sizes, int n_in,
                              void* d_out, int out_size, void* d_ws, size_t ws_size,
                              hipStream_t stream) {
  (void)in_sizes; (void)n_in; (void)out_size; (void)ws_size;
  const float* y = (const float*)d_in[0];
  const float* h = (const float*)d_in[1];
  const float* W = (const float*)d_in[2];
  float* out = (float*)d_out;

  char* base = (char*)d_ws;
  size_t off = 0;
  auto carve = [&](size_t bytes) -> void* {
    void* r = base + off;
    off += (bytes + 255) & ~(size_t)255;
    return r;
  };
  _Float16* y_h  = (_Float16*)carve((size_t)N_DIM * K_DIM * 2);
  _Float16* h_h  = (_Float16*)carve((size_t)M_DIM * H_DIM * 2);
  _Float16* wt_h = (_Float16*)carve((size_t)K_DIM * H_DIM * 2);
  _Float16* q_h  = (_Float16*)carve((size_t)N_DIM * H_DIM * 2);
  _Float16* hdt  = (_Float16*)carve((size_t)H_DIM * M_DIM * 2);
  float* pmax = (float*)carve((size_t)NCHUNK * M_DIM * 4);
  float* psum = (float*)carve((size_t)NCHUNK * M_DIM * 4);
  float* Mb   = (float*)carve((size_t)M_DIM * 4);
  float* Dinv = (float*)carve((size_t)M_DIM * 4);

  k_cvt<<<(N_DIM * K_DIM) / 256, 256, 0, stream>>>(y, y_h, N_DIM * K_DIM);
  k_cvt<<<(M_DIM * H_DIM) / 256, 256, 0, stream>>>(h, h_h, M_DIM * H_DIM);

  dim3 tb(32, 8);
  // Wt[j][k] = W[k][j]
  k_transpose<<<dim3(K_DIM / 32, H_DIM / 32), tb, 0, stream>>>(W, wt_h, K_DIM, H_DIM, nullptr);

  k_gemm_q<<<dim3(N_DIM / 128, H_DIM / 128), 256, 0, stream>>>(y_h, wt_h, q_h);

  k_stats<<<dim3(M_DIM / 128, N_DIM / CHUNK), 256, 0, stream>>>(q_h, h_h, pmax, psum);
  k_reduce<<<M_DIM / 256, 256, 0, stream>>>(pmax, psum, Mb, Dinv);

  // hdt[j][m] = h[m][j] / D[m]
  k_transpose<<<dim3(M_DIM / 32, H_DIM / 32), tb, 0, stream>>>(h, hdt, M_DIM, H_DIM, Dinv);

  k_final<<<N_DIM / 64, 256, 0, stream>>>(q_h, h_h, hdt, Mb, out);
}
